// GConvLSTMModel_49529562857565
// MI455X (gfx1250) — matrix operand order repaired and checked by reading, not compiled, not run on hardware
//
#include <hip/hip_runtime.h>
#include <hip/hip_bf16.h>

typedef __attribute__((ext_vector_type(16))) _Float16 v16h;
typedef __attribute__((ext_vector_type(8)))  _Float16 v8h;
typedef __attribute__((ext_vector_type(8)))  float    v8f;

#define NROWS 100000
#define NPAD  100032       // 1563 * 64 (zero-padded rows in packed A)
#define NBLK  1563
#define FDIM  128
#define HDIM  256
#define KDIM  384          // FDIM + HDIM
#define KTILES 12          // 384 / 32
#define CTILES 64          // 1024 / 16 (4 gates * 256 cols)

// ---------------------------------------------------------------------------
// pack_a: A[n, 0:128) = f16(x[n,:]); A[n, 128:384) = f16(h[n,:]); rows
// [NROWS, NPAD) are zero so the 64-row main blocks never read garbage.
// ---------------------------------------------------------------------------
__global__ void pack_a_kernel(const float* __restrict__ x,
                              const float* __restrict__ h,
                              _Float16* __restrict__ A) {
    int n = blockIdx.x;
    int k = threadIdx.x;                 // 0..383
    float v = 0.0f;
    if (n < NROWS) {
        v = (k < FDIM) ? __builtin_nontemporal_load(&x[(size_t)n * FDIM + k])
                       : __builtin_nontemporal_load(&h[(size_t)n * HDIM + (k - FDIM)]);
    }
    A[(size_t)n * KDIM + k] = (_Float16)v;
}

// ---------------------------------------------------------------------------
// pack_w: pre-swizzle Wbig[K=384][C=1024] into WMMA B-fragment order:
//   Wp[((kt*64 + ct)*32 + lane)*16 + e]  holds element (K, C) with
//   K = kt*32 + (lane<16 ? 0 : 8) + (e<8 ? e : 8 + e),  C = ct*16 + (lane & 15)
// Column C = g*256 + hcol; K<128 -> Wx[g][K][hcol], else Wh[g][K-128][hcol].
// ---------------------------------------------------------------------------
__global__ void pack_w_kernel(const float* __restrict__ Wx,
                              const float* __restrict__ Wh,
                              _Float16* __restrict__ Wp) {
    int idx = blockIdx.x * blockDim.x + threadIdx.x;   // < 393216
    int e    = idx & 15;
    int lane = (idx >> 4) & 31;
    int ct   = (idx >> 9) & 63;
    int kt   = idx >> 15;
    int K = kt * 32 + ((lane & 16) ? 8 : 0) + ((e < 8) ? e : 8 + e);
    int C = ct * 16 + (lane & 15);
    int g = C >> 8;
    int hc = C & 255;
    float v = (K < FDIM)
        ? Wx[((size_t)g * FDIM + K) * HDIM + hc]
        : Wh[((size_t)g * HDIM + (K - FDIM)) * HDIM + hc];
    Wp[idx] = (_Float16)v;
}

// ---------------------------------------------------------------------------
// Main fused kernel: 64 rows per block (4 row-tiles per wave), 8 waves; wave
// w owns hidden cols [w*32, w*32+32) across all 4 gates. The 48 KB A tile is
// async-copied into LDS once per block (ASYNCcnt path), B fragments are
// loaded once per k-step and reused across 4 row-tiles (4x less L2 traffic).
// 384 v_wmma per wave, then in-register LSTM elementwise with NT streaming
// IO + shfl/ds_add_f32 reduction for the final Linear(H,1).
// ---------------------------------------------------------------------------
__global__ __launch_bounds__(256)
void gconv_lstm_main(const _Float16* __restrict__ A,
                     const _Float16* __restrict__ Wp,
                     const float* __restrict__ c_in,
                     const float* __restrict__ bx,
                     const float* __restrict__ bh,
                     const float* __restrict__ bg,
                     const float* __restrict__ wc,
                     const float* __restrict__ Wfc,
                     const float* __restrict__ bfc,
                     float* __restrict__ out,
                     float* __restrict__ hout,
                     float* __restrict__ cout) {
    __shared__ _Float16 shA[64 * KDIM];      // 48 KB staged A tile
    __shared__ float red[64];

    const int tid  = threadIdx.x;
    const int w    = tid >> 5;
    const int lane = tid & 31;
    const int row0 = blockIdx.x * 64;

    if (tid < 64) red[tid] = 0.0f;

    // ---- async copy of the contiguous 48 KB A tile: global -> LDS ----
    {
        const char* gsrc = (const char*)(A + (size_t)row0 * KDIM);
        unsigned ldsbase = (unsigned)(uintptr_t)(&shA[0]);
#pragma unroll
        for (int j = 0; j < 12; ++j) {
            unsigned off = (unsigned)(j * 256 + tid) * 16u;
            unsigned long long gaddr = (unsigned long long)(uintptr_t)(gsrc + off);
            unsigned laddr = ldsbase + off;
            asm volatile("global_load_async_to_lds_b128 %0, %1, off"
                         :: "v"(laddr), "v"(gaddr) : "memory");
        }
        asm volatile("s_wait_asynccnt 0x0" ::: "memory");
    }
    __syncthreads();

    const int m     = lane & 15;               // A-fragment row (within tile)
    const int kbase = (lane & 16) ? 8 : 0;     // A-fragment K sub-offset
    const int n_    = lane & 15;               // C/D-fragment column
    const int m_off = (lane & 16) ? 8 : 0;     // C/D-fragment row offset

    v8f acc[4][2][4] = {};                     // [gate][half][row-tile]

#pragma unroll
    for (int kt = 0; kt < KTILES; ++kt) {
        v16h a[4];
#pragma unroll
        for (int rt = 0; rt < 4; ++rt) {
            const _Float16* ap = &shA[(rt * 16 + m) * KDIM + kt * 32 + kbase];
            union { v16h v; v8h h2[2]; } au;
            au.h2[0] = *(const v8h*)(ap);        // K = kbase + 0..7
            au.h2[1] = *(const v8h*)(ap + 16);   // K = 16 + kbase + 0..7
            a[rt] = au.v;
        }
#pragma unroll
        for (int g = 0; g < 4; ++g) {
#pragma unroll
            for (int t = 0; t < 2; ++t) {
                int ct = g * 16 + w * 2 + t;
                const v16h b = *(const v16h*)(Wp +
                    ((size_t)(kt * CTILES + ct) * 32 + lane) * 16);
#pragma unroll
                for (int rt = 0; rt < 4; ++rt) {
                    acc[g][t][rt] = __builtin_amdgcn_wmma_f32_16x16x32_f16(
                        false, a[rt], false, b, (short)0, acc[g][t][rt],
                        false, false);
                }
            }
        }
    }

    // -------------------- in-register LSTM elementwise --------------------
    float part[4][8];
#pragma unroll
    for (int rt = 0; rt < 4; ++rt)
#pragma unroll
        for (int r = 0; r < 8; ++r) part[rt][r] = 0.0f;

#pragma unroll
    for (int t = 0; t < 2; ++t) {
        const int hc = w * 32 + t * 16 + n_;
        const float bi  = bx[0*HDIM+hc] + bh[0*HDIM+hc] + bg[0*HDIM+hc];
        const float bff = bx[1*HDIM+hc] + bh[1*HDIM+hc] + bg[1*HDIM+hc];
        const float bcc = bx[2*HDIM+hc] + bh[2*HDIM+hc] + bg[2*HDIM+hc];
        const float bo  = bx[3*HDIM+hc] + bh[3*HDIM+hc] + bg[3*HDIM+hc];
        const float w0 = wc[0*HDIM+hc];
        const float w1 = wc[1*HDIM+hc];
        const float w2 = wc[2*HDIM+hc];
        const float wf = Wfc[hc];
#pragma unroll
        for (int rt = 0; rt < 4; ++rt) {
#pragma unroll
            for (int r = 0; r < 8; ++r) {
                const int row = row0 + rt * 16 + m_off + r;
                const bool valid = (row < NROWS);
                const size_t off = (size_t)row * HDIM + hc;
                const float cold = valid ? __builtin_nontemporal_load(&c_in[off])
                                         : 0.0f;
                const float pi = acc[0][t][rt][r] + bi  + w0 * cold;
                const float pf = acc[1][t][rt][r] + bff + w1 * cold;
                const float pc = acc[2][t][rt][r] + bcc;
                const float po = acc[3][t][rt][r] + bo;
                const float ig = 1.0f / (1.0f + __expf(-pi));
                const float fg = 1.0f / (1.0f + __expf(-pf));
                const float cn = fg * cold + ig * tanhf(pc);
                const float og = 1.0f / (1.0f + __expf(-(po + w2 * cn)));
                const float hn = og * tanhf(cn);
                if (valid) {
                    __builtin_nontemporal_store(hn, &hout[off]);
                    __builtin_nontemporal_store(cn, &cout[off]);
                }
                part[rt][r] += fmaxf(hn, 0.0f) * wf;
            }
        }
    }

    // ---- reduce relu(h_new)·Wfc over hidden dim: shfl within 16-lane half,
    // ---- then ds_add_f32 across waves/halves ----
#pragma unroll
    for (int rt = 0; rt < 4; ++rt) {
#pragma unroll
        for (int r = 0; r < 8; ++r) {
            float v = part[rt][r];
            v += __shfl_xor(v, 1, 32);
            v += __shfl_xor(v, 2, 32);
            v += __shfl_xor(v, 4, 32);
            v += __shfl_xor(v, 8, 32);
            if (n_ == 0) atomicAdd(&red[rt * 16 + m_off + r], v);
        }
    }
    __syncthreads();
    if (tid < 64 && (row0 + tid) < NROWS)
        out[row0 + tid] = red[tid] + bfc[0];
}

// ---------------------------------------------------------------------------
extern "C" void kernel_launch(void* const* d_in, const int* in_sizes, int n_in,
                              void* d_out, int out_size, void* d_ws, size_t ws_size,
                              hipStream_t stream) {
    (void)in_sizes; (void)n_in; (void)out_size; (void)ws_size;
    const float* x   = (const float*)d_in[0];
    // d_in[1] = edge_index, d_in[2] = edge_weight : dead (ChebConv K=1)
    const float* h   = (const float*)d_in[3];
    const float* c   = (const float*)d_in[4];
    const float* Wx  = (const float*)d_in[5];
    const float* bx  = (const float*)d_in[6];
    const float* Wh  = (const float*)d_in[7];
    const float* bh  = (const float*)d_in[8];
    const float* wc  = (const float*)d_in[9];
    const float* bg  = (const float*)d_in[10];
    const float* Wfc = (const float*)d_in[11];
    const float* bfc = (const float*)d_in[12];

    _Float16* A  = (_Float16*)d_ws;                                    // 76.8 MB
    _Float16* Wp = (_Float16*)((char*)d_ws + (size_t)NPAD * KDIM * 2); // 768 KB

    float* out  = (float*)d_out;
    float* hout = out + NROWS;
    float* cout = hout + (size_t)NROWS * HDIM;

    pack_a_kernel<<<NPAD, KDIM, 0, stream>>>(x, h, A);
    pack_w_kernel<<<(KTILES * CTILES * 32 * 16) / 256, 256, 0, stream>>>(Wx, Wh, Wp);
    gconv_lstm_main<<<NBLK, 256, 0, stream>>>(A, Wp, c, bx, bh, bg,
                                              wc, Wfc, bfc, out, hout, cout);
}
